// GCNEncoder_34600256537276
// MI455X (gfx1250) — matrix operand order repaired and checked by reading, not compiled, not run on hardware
//
#include <hip/hip_runtime.h>

// ---------------------------------------------------------------------------
// 2-layer GCN for MI455X (gfx1250):
//   deg/rsqrt -> [GEMM bf16-WMMA, ping-pong pipelined, dinv-scaled] ->
//   self-loop copy -> edge scatter (fp32 atomics, ~L2-resident) -> bias+ReLU
// ---------------------------------------------------------------------------

#define N_NODES 100000
#define F_IN    512
#define F_HID   512
#define F_OUT   256

typedef __attribute__((ext_vector_type(16))) __bf16 v16bf;
typedef __attribute__((ext_vector_type(8)))  float  v8f;

__device__ __forceinline__ unsigned short f2bf(float f) {
  // round-to-nearest-even fp32 -> bf16 (scalar pack path)
  unsigned int u = __float_as_uint(f);
  u += 0x7FFFu + ((u >> 16) & 1u);
  return (unsigned short)(u >> 16);
}

// ---------------------------------------------------------------------------
// small helpers
// ---------------------------------------------------------------------------
__global__ void k_fill(float* __restrict__ p, float v, int n) {
  int i = blockIdx.x * blockDim.x + threadIdx.x;
  if (i < n) p[i] = v;
}

__global__ void k_deg(const int* __restrict__ dst, float* __restrict__ deg, int E) {
  int i = blockIdx.x * blockDim.x + threadIdx.x;
  if (i < E) atomicAdd(&deg[dst[i]], 1.0f);
}

__global__ void k_rsqrt(float* __restrict__ d, int n) {
  int i = blockIdx.x * blockDim.x + threadIdx.x;
  if (i < n) d[i] = rsqrtf(d[i]);   // deg >= 1 always (self-loop)
}

__global__ void k_copy4(const float4* __restrict__ s, float4* __restrict__ d, long n4) {
  long i = (long)blockIdx.x * blockDim.x + threadIdx.x;
  if (i < n4) d[i] = s[i];
}

// ---------------------------------------------------------------------------
// pack fp32 W[K][Nout] (row-major) into WMMA B-fragment tile order (bf16):
// tile (kt,nt) = 32x16, 512 bf16 per tile; lane L owns 16 contiguous bf16.
// element (k0 = k&31, n0 = n&15): lane = n0 + 16*(k0>>4), pos = k0&15
// ---------------------------------------------------------------------------
__global__ void k_packW(const float* __restrict__ W, unsigned short* __restrict__ Wp,
                        int K, int Nout) {
  int tid = blockIdx.x * blockDim.x + threadIdx.x;
  if (tid >= K * Nout) return;
  int k = tid / Nout, n = tid % Nout;
  int kt = k >> 5, nt = n >> 4;
  int k0 = k & 31, n0 = n & 15;
  int L = n0 + ((k0 >> 4) << 4);
  int p = k0 & 15;
  Wp[((size_t)(kt * (Nout >> 4) + nt) << 9) + L * 16 + p] = f2bf(W[tid]);
}

// ---------------------------------------------------------------------------
// GEMM: Out[row, col] = dinv[row] * sum_k X[row,k] * W[k,col]
// one wave per 16x64 output tile; K loop in steps of 32; bf16 WMMA, f32 acc.
// Unroll-by-2 ping-pong buffering: no rotation copies, loads overlap WMMAs.
// ---------------------------------------------------------------------------
__device__ __forceinline__ void loadA4(const float* __restrict__ xrow, int kb,
                                       float4* __restrict__ a4) {
  const float4* p = (const float4*)(xrow + kb);        // K chunk [kb .. kb+7]
  const float4* q = (const float4*)(xrow + kb + 8);    // K chunk [kb+8 .. kb+15]
  a4[0] = p[0]; a4[1] = p[1];
  a4[2] = q[0]; a4[3] = q[1];
}

__device__ __forceinline__ void loadB4(const unsigned short* __restrict__ wt,
                                       v16bf* __restrict__ b) {
#pragma unroll
  for (int t = 0; t < 4; ++t) b[t] = *(const v16bf*)(wt + t * 512);
}

__device__ __forceinline__ v16bf cvtA(const float4* __restrict__ a4) {
  const float* f = (const float*)a4;
  v16bf r;
#pragma unroll
  for (int p = 0; p < 16; ++p) r[p] = (__bf16)f[p];    // v_cvt_pk_bf16_f32
  return r;
}

__device__ __forceinline__ void wmma4(const float4* __restrict__ a4,
                                      const v16bf* __restrict__ b,
                                      v8f& c0, v8f& c1, v8f& c2, v8f& c3) {
  v16bf af = cvtA(a4);
  c0 = __builtin_amdgcn_wmma_f32_16x16x32_bf16(false, af, false, b[0], (short)0, c0, false, false);
  c1 = __builtin_amdgcn_wmma_f32_16x16x32_bf16(false, af, false, b[1], (short)0, c1, false, false);
  c2 = __builtin_amdgcn_wmma_f32_16x16x32_bf16(false, af, false, b[2], (short)0, c2, false, false);
  c3 = __builtin_amdgcn_wmma_f32_16x16x32_bf16(false, af, false, b[3], (short)0, c3, false, false);
}

__global__ __launch_bounds__(32)
void k_gemm_bf16(const float* __restrict__ X,
                 const unsigned short* __restrict__ Wp,
                 const float* __restrict__ dinv,
                 float* __restrict__ Out,
                 int K, int Nout) {
  const int lane = threadIdx.x & 31;
  const int half = lane >> 4;
  const int l    = lane & 15;
  const int rowBase = blockIdx.x << 4;
  const int colBase = blockIdx.y << 6;
  const int ntiles  = Nout >> 4;

  const float* __restrict__ xrow = X + (size_t)(rowBase + l) * K;
  const unsigned short* __restrict__ wbase =
      Wp + (size_t)(colBase >> 4) * 512 + lane * 16;   // B tile base for kt=0
  const size_t kStride = (size_t)ntiles << 9;          // elements per K-step of tiles
  const int ao = half << 4;                            // per-lane A offset within chunk

  v8f c0 = {}, c1 = {}, c2 = {}, c3 = {};

  float4 a0[4], a1[4];
  v16bf  b0[4], b1[4];

  // prologue: loads for kk = 0                   (K assumed >= 64, K/32 even)
  loadA4(xrow, ao, a0);
  loadB4(wbase, b0);

  int kk = 0;
  for (; kk < K - 64; kk += 64) {
    // phase 0: issue kk+32 loads, compute kk
    loadA4(xrow, kk + 32 + ao, a1);
    loadB4(wbase + (size_t)((kk >> 5) + 1) * kStride, b1);
    __builtin_prefetch(xrow + kk + 64 + ao, 0, 0);     // global_prefetch
    wmma4(a0, b0, c0, c1, c2, c3);
    // phase 1: issue kk+64 loads, compute kk+32
    loadA4(xrow, kk + 64 + ao, a0);
    loadB4(wbase + (size_t)((kk >> 5) + 2) * kStride, b0);
    wmma4(a1, b1, c0, c1, c2, c3);
  }
  // peeled final pair: kk = K-64
  loadA4(xrow, kk + 32 + ao, a1);
  loadB4(wbase + (size_t)((kk >> 5) + 1) * kStride, b1);
  wmma4(a0, b0, c0, c1, c2, c3);
  wmma4(a1, b1, c0, c1, c2, c3);

  // C/D layout: VGPR r -> (M = r + 8*half, N = l)
#pragma unroll
  for (int r = 0; r < 8; ++r) {
    const int row = rowBase + r + (half << 3);
    const float sc = dinv[row];
    float* o = Out + (size_t)row * Nout + colBase + l;
    o[0]  = c0[r] * sc;
    o[16] = c1[r] * sc;
    o[32] = c2[r] * sc;
    o[48] = c3[r] * sc;
  }
}

// ---------------------------------------------------------------------------
// edge scatter: acc[dst,:] += hs[src,:]   (one wave per edge, float4 gathers,
// fp32 atomics; dest array ~fits 192MB L2 so atomics resolve on-chip)
// ---------------------------------------------------------------------------
__global__ __launch_bounds__(256)
void k_spmm_edges(const int* __restrict__ esrc, const int* __restrict__ edst,
                  const float* __restrict__ hs, float* __restrict__ acc,
                  int E, int F) {
  int wid  = blockIdx.x * (blockDim.x >> 5) + (threadIdx.x >> 5);
  int lane = threadIdx.x & 31;
  if (wid >= E) return;
  int s = esrc[wid], d = edst[wid];
  const float4* __restrict__ h4 = (const float4*)(hs + (size_t)s * F);
  float*        __restrict__ ar = acc + (size_t)d * F;
  const int F4 = F >> 2;
  for (int i = lane; i < F4; i += 32) {
    float4 v = h4[i];
    float* ap = ar + i * 4;
    atomicAdd(ap + 0, v.x);
    atomicAdd(ap + 1, v.y);
    atomicAdd(ap + 2, v.z);
    atomicAdd(ap + 3, v.w);
  }
}

// out[i,f] = relu(dinv[i] * acc[i,f] + b[f]);  F = 1<<fshift
__global__ void k_bias_relu(const float* __restrict__ acc, const float* __restrict__ dinv,
                            const float* __restrict__ b, float* __restrict__ out,
                            long total, int fshift) {
  long i = (long)blockIdx.x * blockDim.x + threadIdx.x;
  if (i >= total) return;
  int row = (int)(i >> fshift);
  int f   = (int)(i & ((1 << fshift) - 1));
  out[i] = fmaxf(fmaf(dinv[row], acc[i], b[f]), 0.0f);
}

// ---------------------------------------------------------------------------
extern "C" void kernel_launch(void* const* d_in, const int* in_sizes, int n_in,
                              void* d_out, int out_size, void* d_ws, size_t ws_size,
                              hipStream_t stream) {
  const float* x  = (const float*)d_in[0];
  const int*   ei = (const int*)d_in[1];     // [2,E] flat: src then dst
  const float* W1 = (const float*)d_in[2];
  const float* b1 = (const float*)d_in[3];
  const float* W2 = (const float*)d_in[4];
  const float* b2 = (const float*)d_in[5];
  float* out = (float*)d_out;

  const int N = N_NODES;
  const int E = in_sizes[1] / 2;

  char* ws = (char*)d_ws;
  float*          dinv = (float*)(ws);                                   // N f32 (also deg)
  unsigned short* W1p  = (unsigned short*)(ws + (1u << 19));             // 512KB off
  unsigned short* W2p  = (unsigned short*)(ws + (1u << 19) + 512*512*2);
  float* bufA = (float*)(ws + (2u << 20));                               // N*512 f32
  float* bufB = (float*)((char*)bufA + (size_t)N * F_HID * 4);           // N*512 f32
  float* hs2  = bufB;                                                    // N*256 f32
  float* acc2 = (float*)((char*)bufB + (size_t)N * F_OUT * 4);           // N*256 f32

  const dim3 b256(256);

  // ---- normalization: deg = 1 + in-degree; dinv = rsqrt(deg)
  k_fill  <<<(N + 255) / 256, b256, 0, stream>>>(dinv, 1.0f, N);
  k_deg   <<<(E + 255) / 256, b256, 0, stream>>>(ei + E, dinv, E);
  k_rsqrt <<<(N + 255) / 256, b256, 0, stream>>>(dinv, N);

  // ---- pack weights to bf16 WMMA B-fragment tiles
  k_packW <<<(512 * 512 + 255) / 256, b256, 0, stream>>>(W1, W1p, 512, F_HID);
  k_packW <<<(512 * 256 + 255) / 256, b256, 0, stream>>>(W2, W2p, 512, F_OUT);

  // ---- layer 1: hs1 = (x @ W1) * dinv[row]   (bufA)
  k_gemm_bf16 <<<dim3(N / 16, F_HID / 64), dim3(32), 0, stream>>>(x, W1p, dinv, bufA, F_IN, F_HID);
  // acc1 = hs1 (self-loop term), then scatter edges, then finalize -> G1 (bufA)
  long n4a = (long)N * F_HID / 4;
  k_copy4     <<<(unsigned)((n4a + 255) / 256), b256, 0, stream>>>((const float4*)bufA, (float4*)bufB, n4a);
  k_spmm_edges<<<(E + 7) / 8, b256, 0, stream>>>(ei, ei + E, bufA, bufB, E, F_HID);
  k_bias_relu <<<(unsigned)(((long)N * F_HID + 255) / 256), b256, 0, stream>>>(bufB, dinv, b1, bufA, (long)N * F_HID, 9);

  // ---- layer 2: hs2 = (G1 @ W2) * dinv[row]
  k_gemm_bf16 <<<dim3(N / 16, F_OUT / 64), dim3(32), 0, stream>>>(bufA, W2p, dinv, hs2, F_HID, F_OUT);
  long n4b = (long)N * F_OUT / 4;
  k_copy4     <<<(unsigned)((n4b + 255) / 256), b256, 0, stream>>>((const float4*)hs2, (float4*)acc2, n4b);
  k_spmm_edges<<<(E + 7) / 8, b256, 0, stream>>>(ei, ei + E, hs2, acc2, E, F_OUT);
  k_bias_relu <<<(unsigned)(((long)N * F_OUT + 255) / 256), b256, 0, stream>>>(acc2, dinv, b2, out, (long)N * F_OUT, 8);
}
